// LSTMEncoderDecoderWrapper_19413252178214
// MI455X (gfx1250) — matrix operand order repaired and checked by reading, not compiled, not run on hardware
//
#include <hip/hip_runtime.h>

// ---- problem dims ----
#define BB   256
#define SS   512
#define TT   48
#define DIN  32
#define HH   512
#define FF   32
#define UU   128
#define G4H  2048          // 4*H
#define KENC 544           // H + DIN        (encoder fused GEMM K)
#define KDEC 1056          // (H+F) + H      (decoder fused GEMM K)
#define HF   544           // H + F  (offset of h slot in decoder input)

typedef __attribute__((ext_vector_type(16))) __bf16 v16bf;
typedef __attribute__((ext_vector_type(8)))  __bf16 v8bf;
typedef __attribute__((ext_vector_type(2)))  __bf16 v2bf;
typedef __attribute__((ext_vector_type(8)))  float  v8f;

__device__ __forceinline__ float sigm(float x) { return 1.0f / (1.0f + __expf(-x)); }

__device__ __forceinline__ v16bf make_frag(const __bf16* lo, const __bf16* hi)
{
    v8bf l = *(const v8bf*)lo;
    v8bf h = *(const v8bf*)hi;
    v16bf r;
#pragma unroll
    for (int i = 0; i < 8; ++i) { r[i] = l[i]; r[i + 8] = h[i]; }
    return r;
}

// =====================================================================
// Fused LSTM step:  gates = A * W^T + bias, then cell update, all in one
// kernel. Wave w owns batch-tile tm (16 rows) and hidden-tile tj (16 j's)
// and computes the FOUR gate tiles (j, j+H, j+2H, j+3H) with a shared A
// fragment -> i/f/g/o live in the wave's accumulators, so the sigmoid/
// tanh cell update is done in-register (no gates buffer round-trip).
// Grid: 512 waves = 64 blocks x 256 threads (M=256, Htiles=32 fixed).
// Reads A (bf16 packed input, double buffered), writes h (bf16) into the
// next step's packed input buffer(s).
// =====================================================================
__global__ __launch_bounds__(256)
void lstm_step_fused(const __bf16* __restrict__ A, int lda,    // [B][K] cur input
                     const __bf16* __restrict__ W, int ldw,    // [4H][K] packed weights
                     const float*  __restrict__ bias,          // [4H]
                     float* __restrict__ cst,                  // [B][H] cell state
                     int K,
                     __bf16* __restrict__ h_dst, int h_ld, int h_off,   // next input buf
                     __bf16* __restrict__ eout_t,              // enc_out + t*H (or null)
                     const float* __restrict__ x_next,         // x + (t+1)*DIN (or null)
                     __bf16* __restrict__ x_dst,               // next enc buf (or null)
                     __bf16* __restrict__ h_dst2, int h2_ld, int h2_off) // optional 2nd dest
{
    const int lane = threadIdx.x & 31;
    const int wave = threadIdx.x >> 5;
    const int grp  = blockIdx.x * 8 + wave;      // 0..511
    const int tm   = grp >> 5;                   // batch tile (16 rows)
    const int tj   = grp & 31;                   // hidden tile (16 cols)
    const int hl   = lane >> 4;
    const int l15  = lane & 15;
    const int j    = tj * 16 + l15;

    const __bf16* arow = A + (long)(tm * 16 + l15) * lda + hl * 8;
    const __bf16* brow = W + (long)(j) * ldw + hl * 8;

    v8f acc[4] = {};
    for (int k = 0; k < K; k += 32) {
        v16bf a = make_frag(arow + k, arow + k + 16);
#pragma unroll
        for (int g = 0; g < 4; ++g) {
            const __bf16* br = brow + (long)(g * HH) * ldw + k;
            v16bf b = make_frag(br, br + 16);
            acc[g] = __builtin_amdgcn_wmma_f32_16x16x32_bf16(
                         false, a, false, b, (short)0, acc[g], false, false);
        }
    }

    const float bi = bias[j];
    const float bf_ = bias[j + HH];
    const float bg = bias[j + 2 * HH];
    const float bo = bias[j + 3 * HH];

#pragma unroll
    for (int r = 0; r < 8; ++r) {
        const int m = tm * 16 + hl * 8 + r;      // batch row
        float gi = acc[0][r] + bi;
        float gf = acc[1][r] + bf_;
        float gg = acc[2][r] + bg;
        float go = acc[3][r] + bo;
        float cv = cst[m * HH + j];
        float cn = sigm(gf) * cv + sigm(gi) * tanhf(gg);
        float hn = sigm(go) * tanhf(cn);
        cst[m * HH + j] = cn;
        __bf16 hb = (__bf16)hn;
        h_dst[(long)m * h_ld + h_off + j] = hb;
        if (eout_t)  eout_t[(long)m * SS * HH + j] = hb;
        if (h_dst2)  h_dst2[(long)m * h2_ld + h2_off + j] = hb;
    }

    // stage x_{t+1} into the next encoder input buffer (B*DIN = 8192 elems)
    if (x_dst && x_next) {
        int idx = blockIdx.x * 256 + threadIdx.x;
        if (idx < BB * DIN) {
            int b2 = idx >> 5, f2 = idx & 31;
            x_dst[b2 * KENC + HH + f2] = (__bf16)x_next[(long)b2 * SS * DIN + f2];
        }
    }
}

// =====================================================================
// Generic bf16 WMMA GEMM, 4 consecutive N-tiles per wave (A reuse):
//   C[M x N] = A[M x K] * W[N x K]^T + bias[N]
// Launch with (M/16 * N/64)/8 blocks of 256 threads.
// =====================================================================
__global__ __launch_bounds__(256)
void gemm_bf16_wmma_x4(const __bf16* __restrict__ A, int lda,
                       const __bf16* __restrict__ W, int ldw,
                       const float*  __restrict__ bias,
                       float* __restrict__ C, int ldc,
                       int groups_n, int K)
{
    const int lane = threadIdx.x & 31;
    const int wave = threadIdx.x >> 5;
    const int grp  = blockIdx.x * 8 + wave;
    const int tm   = grp / groups_n;
    const int tg   = grp - tm * groups_n;
    const int n0   = tg * 64;
    const int hl   = lane >> 4;
    const int l15  = lane & 15;

    const __bf16* arow = A + (long)(tm * 16 + l15) * lda + hl * 8;
    const __bf16* brow = W + (long)(n0 + l15) * ldw + hl * 8;

    v8f acc[4] = {};
    for (int k = 0; k < K; k += 32) {
        v16bf a = make_frag(arow + k, arow + k + 16);
#pragma unroll
        for (int t = 0; t < 4; ++t) {
            const __bf16* br = brow + (long)(t * 16) * ldw + k;
            v16bf b = make_frag(br, br + 16);
            acc[t] = __builtin_amdgcn_wmma_f32_16x16x32_bf16(
                         false, a, false, b, (short)0, acc[t], false, false);
        }
    }

#pragma unroll
    for (int t = 0; t < 4; ++t) {
        const float bv = bias ? bias[n0 + t * 16 + l15] : 0.0f;
        float* crow = C + (long)(tm * 16 + hl * 8) * ldc + n0 + t * 16 + l15;
#pragma unroll
        for (int r = 0; r < 8; ++r)
            crow[(long)r * ldc] = acc[t][r] + bv;
    }
}

// =====================================================================
// Single-tile WMMA GEMM (for small N, e.g. the 32-wide output projection)
// Launch with (M/16 * N/16)/8 blocks of 256 threads.
// =====================================================================
__global__ __launch_bounds__(256)
void gemm_bf16_wmma(const __bf16* __restrict__ A, int lda,
                    const __bf16* __restrict__ W, int ldw,
                    const float*  __restrict__ bias,
                    float* __restrict__ C, int ldc,
                    int tiles_n, int K)
{
    const int lane = threadIdx.x & 31;
    const int wave = threadIdx.x >> 5;
    const int tile = blockIdx.x * 8 + wave;
    const int tm   = tile / tiles_n;
    const int tn   = tile - tm * tiles_n;
    const int hl   = lane >> 4;
    const int l15  = lane & 15;

    const __bf16* arow = A + (long)(tm * 16 + l15) * lda + hl * 8;
    const __bf16* brow = W + (long)(tn * 16 + l15) * ldw + hl * 8;

    v8f acc = {};
    for (int k = 0; k < K; k += 32) {
        v16bf a = make_frag(arow + k, arow + k + 16);
        v16bf b = make_frag(brow + k, brow + k + 16);
        acc = __builtin_amdgcn_wmma_f32_16x16x32_bf16(
                  false, a, false, b, (short)0, acc, false, false);
    }
    const float bv = bias ? bias[tn * 16 + l15] : 0.0f;
    float* crow = C + (long)(tm * 16 + hl * 8) * ldc + tn * 16 + l15;
#pragma unroll
    for (int r = 0; r < 8; ++r)
        crow[(long)r * ldc] = acc[r] + bv;
}

// =====================================================================
// Weight packing / conversion kernels (run once)
// =====================================================================
__global__ void cvt_bf16_kernel(__bf16* __restrict__ dst, const float* __restrict__ src, int n)
{
    int i = blockIdx.x * 256 + threadIdx.x;
    if (i < n) dst[i] = (__bf16)src[i];
}

__global__ void pack_enc_w_kernel(__bf16* __restrict__ dst,
                                  const float* __restrict__ Whh,
                                  const float* __restrict__ Wih)
{
    int i = blockIdx.x * 256 + threadIdx.x;   // 2048*544 exact
    int n = i / KENC, k = i - n * KENC;
    float v = (k < HH) ? Whh[n * HH + k] : Wih[n * DIN + (k - HH)];
    dst[i] = (__bf16)v;
}

__global__ void pack_dec_w_kernel(__bf16* __restrict__ dst,
                                  const float* __restrict__ Wih,
                                  const float* __restrict__ Whh)
{
    int i = blockIdx.x * 256 + threadIdx.x;   // 2048*1056 exact
    int n = i / KDEC, k = i - n * KDEC;
    float v = (k < HF) ? Wih[n * HF + k] : Whh[n * HH + (k - HF)];
    dst[i] = (__bf16)v;
}

__global__ void bias_sum_kernel(float* __restrict__ dst,
                                const float* __restrict__ a,
                                const float* __restrict__ b, int n)
{
    int i = blockIdx.x * 256 + threadIdx.x;
    if (i < n) dst[i] = a[i] + b[i];
}

// =====================================================================
// State init: c=0, enc buf0: h=0 / x = x[:,0,:] ; dec buf0: prev = 0
// =====================================================================
__global__ void init_kernel(const float* __restrict__ x,
                            float* __restrict__ c,
                            __bf16* __restrict__ enc0,
                            __bf16* __restrict__ dec0)
{
    int idx = blockIdx.x * 256 + threadIdx.x;   // B*H exact
    int b = idx >> 9, j = idx & 511;
    c[idx] = 0.0f;
    enc0[b * KENC + j] = (__bf16)0.0f;
    if (j < DIN) {
        enc0[b * KENC + HH + j] = (__bf16)x[(long)b * SS * DIN + j];
        dec0[b * KDEC + HH + j] = (__bf16)0.0f;   // prev-output slot
    }
}

// convert decoder output t into the bf16 "prev" slot of the next input buf
__global__ void prev_kernel(const float* __restrict__ out_t,  // d_out + t*F, stride T*F
                            __bf16* __restrict__ dec_nxt)
{
    int idx = blockIdx.x * 256 + threadIdx.x;  // B*F exact
    int b = idx >> 5, f = idx & 31;
    dec_nxt[b * KDEC + HH + f] = (__bf16)out_t[(long)b * TT * FF + f];
}

// =====================================================================
// Bahdanau attention, one workgroup (256 thr) per batch element.
// Phase 3 streams enc_out (bf16, HBM-bound part) with 4B/lane loads and
// software prefetch (global_prefetch_b8).
// =====================================================================
__global__ __launch_bounds__(256)
void attn_kernel(const float* __restrict__ q,         // [B][U]
                 const float* __restrict__ enc_proj,  // [B*S][U]
                 const __bf16* __restrict__ enc_out,  // [B*S][H]
                 const float* __restrict__ Vw,        // [U]
                 const float* __restrict__ Vb,        // [1]
                 __bf16* __restrict__ dec_cur)        // ctx -> dec_cur[b][0..H)
{
    __shared__ float s_q[UU];
    __shared__ float s_v[UU];
    __shared__ float s_score[SS];
    __shared__ float s_tmp[256];

    const int b    = blockIdx.x;
    const int tid  = threadIdx.x;
    const int lane = tid & 31;
    const int wave = tid >> 5;

    if (tid < UU) { s_q[tid] = q[b * UU + tid]; s_v[tid] = Vw[tid]; }
    __syncthreads();

    // ---- phase 1: scores
    for (int s = wave; s < SS; s += 8) {
        const float* ep = enc_proj + ((long)(b * SS + s)) * UU + lane * 4;
        float p = 0.0f;
#pragma unroll
        for (int u = 0; u < 4; ++u) {
            int uu = lane * 4 + u;
            p += s_v[uu] * tanhf(s_q[uu] + ep[u]);
        }
        p += __shfl_xor(p, 16);
        p += __shfl_xor(p, 8);
        p += __shfl_xor(p, 4);
        p += __shfl_xor(p, 2);
        p += __shfl_xor(p, 1);
        if (lane == 0) s_score[s] = p + Vb[0];
    }
    __syncthreads();

    // ---- phase 2: softmax over S=512
    float m = fmaxf(s_score[tid], s_score[tid + 256]);
    s_tmp[tid] = m; __syncthreads();
    for (int off = 128; off > 0; off >>= 1) {
        if (tid < off) s_tmp[tid] = fmaxf(s_tmp[tid], s_tmp[tid + off]);
        __syncthreads();
    }
    m = s_tmp[0];
    __syncthreads();
    float e0 = __expf(s_score[tid] - m);
    float e1 = __expf(s_score[tid + 256] - m);
    s_tmp[tid] = e0 + e1; __syncthreads();
    for (int off = 128; off > 0; off >>= 1) {
        if (tid < off) s_tmp[tid] += s_tmp[tid + off];
        __syncthreads();
    }
    float inv = 1.0f / s_tmp[0];
    __syncthreads();
    s_score[tid]       = e0 * inv;
    s_score[tid + 256] = e1 * inv;
    __syncthreads();

    // ---- phase 3: context; thread owns h-dims {2*tid, 2*tid+1}
    const v2bf* eo = (const v2bf*)(enc_out + ((long)b * SS) * HH) + tid;
    float acc0 = 0.0f, acc1 = 0.0f;
    for (int s = 0; s < SS; ++s) {
        __builtin_prefetch(eo + (long)(s + 4) * (HH / 2), 0, 0);
        float w = s_score[s];
        v2bf e = eo[(long)s * (HH / 2)];
        acc0 += w * (float)e[0];
        acc1 += w * (float)e[1];
    }
    dec_cur[b * KDEC + 2 * tid]     = (__bf16)acc0;
    dec_cur[b * KDEC + 2 * tid + 1] = (__bf16)acc1;
}

// =====================================================================
// Host driver
// =====================================================================
extern "C" void kernel_launch(void* const* d_in, const int* in_sizes, int n_in,
                              void* d_out, int out_size, void* d_ws, size_t ws_size,
                              hipStream_t stream)
{
    const float* x        = (const float*)d_in[0];
    const float* enc_Wih  = (const float*)d_in[1];
    const float* enc_Whh  = (const float*)d_in[2];
    const float* enc_bih  = (const float*)d_in[3];
    const float* enc_bhh  = (const float*)d_in[4];
    const float* dec_Wih  = (const float*)d_in[5];
    const float* dec_Whh  = (const float*)d_in[6];
    const float* dec_bih  = (const float*)d_in[7];
    const float* dec_bhh  = (const float*)d_in[8];
    const float* att_W1_w = (const float*)d_in[9];
    const float* att_W1_b = (const float*)d_in[10];
    const float* att_W2_w = (const float*)d_in[11];
    const float* att_W2_b = (const float*)d_in[12];
    const float* att_V_w  = (const float*)d_in[13];
    const float* att_V_b  = (const float*)d_in[14];
    const float* out_w    = (const float*)d_in[15];
    const float* out_b    = (const float*)d_in[16];
    float* out = (float*)d_out;

    char* ws = (char*)d_ws;
    size_t off = 0;
    auto carve = [&](size_t bytes) -> char* {
        char* p = ws + off;
        off = (off + bytes + 255) & ~(size_t)255;
        return p;
    };

    __bf16* Wenc    = (__bf16*)carve((size_t)G4H * KENC * 2);   // [2048][544]
    __bf16* Wdec    = (__bf16*)carve((size_t)G4H * KDEC * 2);   // [2048][1056]
    __bf16* W1b     = (__bf16*)carve((size_t)UU * HH * 2);
    __bf16* W2b     = (__bf16*)carve((size_t)UU * HH * 2);
    __bf16* OWb     = (__bf16*)carve((size_t)FF * HH * 2);
    float*  enc_b   = (float*)carve((size_t)G4H * 4);
    float*  dec_b   = (float*)carve((size_t)G4H * 4);
    __bf16* ebuf[2];
    ebuf[0]         = (__bf16*)carve((size_t)BB * KENC * 2);    // [h | x_t] ping
    ebuf[1]         = (__bf16*)carve((size_t)BB * KENC * 2);    // [h | x_t] pong
    __bf16* dbuf[2];
    dbuf[0]         = (__bf16*)carve((size_t)BB * KDEC * 2);    // [ctx | prev | h] ping
    dbuf[1]         = (__bf16*)carve((size_t)BB * KDEC * 2);    // [ctx | prev | h] pong
    float*  cstate  = (float*)carve((size_t)BB * HH * 4);
    float*  qbuf    = (float*)carve((size_t)BB * UU * 4);
    __bf16* enc_out = (__bf16*)carve((size_t)BB * SS * HH * 2); // 134 MB
    float*  enc_prj = (float*)carve((size_t)BB * SS * UU * 4);  // 67 MB
    (void)ws_size; (void)in_sizes; (void)n_in; (void)out_size;

    // ---- one-time packing ----
    pack_enc_w_kernel<<<(G4H * KENC) / 256, 256, 0, stream>>>(Wenc, enc_Whh, enc_Wih);
    pack_dec_w_kernel<<<(G4H * KDEC) / 256, 256, 0, stream>>>(Wdec, dec_Wih, dec_Whh);
    cvt_bf16_kernel<<<(UU * HH) / 256, 256, 0, stream>>>(W1b, att_W1_w, UU * HH);
    cvt_bf16_kernel<<<(UU * HH) / 256, 256, 0, stream>>>(W2b, att_W2_w, UU * HH);
    cvt_bf16_kernel<<<(FF * HH) / 256, 256, 0, stream>>>(OWb, out_w, FF * HH);
    bias_sum_kernel<<<G4H / 256, 256, 0, stream>>>(enc_b, enc_bih, enc_bhh, G4H);
    bias_sum_kernel<<<G4H / 256, 256, 0, stream>>>(dec_b, dec_bih, dec_bhh, G4H);

    init_kernel<<<(BB * HH) / 256, 256, 0, stream>>>(x, cstate, ebuf[0], dbuf[0]);

    // ---- encoder: S fused GEMM+cell steps (ping-pong input buffers) ----
    for (int t = 0; t < SS; ++t) {
        __bf16* cur = ebuf[t & 1];
        __bf16* nxt = ebuf[(t + 1) & 1];
        int last = (t == SS - 1);
        lstm_step_fused<<<64, 256, 0, stream>>>(
            cur, KENC, Wenc, KENC, enc_b, cstate, KENC,
            nxt, KENC, 0,
            enc_out + (size_t)t * HH,
            (t + 1 < SS) ? (x + (size_t)(t + 1) * DIN) : nullptr,
            (t + 1 < SS) ? nxt : nullptr,
            last ? dbuf[0] : nullptr, KDEC, HF);
    }

    // ---- hoisted attention projection: enc_proj = enc_out * W2^T + b2 ----
    // M=131072, N=128 -> 8192 x 2 groups = 16384 waves -> 2048 blocks
    gemm_bf16_wmma_x4<<<2048, 256, 0, stream>>>(enc_out, HH, W2b, HH, att_W2_b,
                                                enc_prj, UU, UU / 64, HH);

    // ---- decoder: T sequential steps ----
    for (int t = 0; t < TT; ++t) {
        __bf16* cur = dbuf[t & 1];
        __bf16* nxt = dbuf[(t + 1) & 1];
        // q = h * W1^T + b1 ; M=256,N=128 -> 16 x 2 groups -> 4 blocks
        gemm_bf16_wmma_x4<<<4, 256, 0, stream>>>(cur + HF, KDEC, W1b, HH,
                                                 att_W1_b, qbuf, UU, UU / 64, HH);
        attn_kernel<<<BB, 256, 0, stream>>>(qbuf, enc_prj, enc_out,
                                            att_V_w, att_V_b, cur);
        // fused gates GEMM + cell: writes h_{t+1} into nxt's h slot
        lstm_step_fused<<<64, 256, 0, stream>>>(
            cur, KDEC, Wdec, KDEC, dec_b, cstate, KDEC,
            nxt, KDEC, HF,
            nullptr, nullptr, nullptr,
            nullptr, 0, 0);
        // out_t = h_new * out_w^T + out_b ; M=256,N=32 -> 32 tiles -> 4 blocks
        gemm_bf16_wmma<<<4, 256, 0, stream>>>(nxt + HF, KDEC, OWb, HH,
                                              out_b, out + (size_t)t * FF, TT * FF,
                                              FF / 16, HH);
        prev_kernel<<<(BB * FF) / 256, 256, 0, stream>>>(out + (size_t)t * FF, nxt);
    }
}
